// Attention_66666482368652
// MI455X (gfx1250) — matrix operand order repaired and checked by reading, not compiled, not run on hardware
//
#include <hip/hip_runtime.h>
#include <hip/hip_bf16.h>

// ---------------------------------------------------------------- types
typedef __bf16 bf16_t;
typedef bf16_t v16bf __attribute__((ext_vector_type(16)));
typedef bf16_t v8bf  __attribute__((ext_vector_type(8)));
typedef float  v8f   __attribute__((ext_vector_type(8)));

union Frag16 { v16bf v; v8bf h[2]; };

__device__ __forceinline__ bf16_t to_bf16(float f) {
    unsigned u = __builtin_bit_cast(unsigned, f);
    u += 0x7FFFu + ((u >> 16) & 1u);                 // round-to-nearest-even
    unsigned short hs = (unsigned short)(u >> 16);
    return __builtin_bit_cast(bf16_t, hs);
}

// A fragment: 16x32 bf16 (MxK) from row-major [M][K] source, stride in bf16 units.
// lanes 0-15 -> K units {0..7, 16..23}; lanes 16-31 -> {8..15, 24..31}.
__device__ __forceinline__ v16bf load_frag_a(const bf16_t* src, int stride, int lane) {
    const int half = lane >> 4, r = lane & 15;
    const bf16_t* p = src + r * stride + half * 8;
    Frag16 f;
    f.h[0] = *(const v8bf*)(p);
    f.h[1] = *(const v8bf*)(p + 16);
    return f.v;
}
// B fragment: 32x16 bf16 (KxN) from row-major [N][K] (column-major B) source.
__device__ __forceinline__ v16bf load_frag_b(const bf16_t* srcT, int stride, int lane) {
    const int half = lane >> 4, n = lane & 15;
    const bf16_t* p = srcT + n * stride + half * 8;
    Frag16 f;
    f.h[0] = *(const v8bf*)(p);
    f.h[1] = *(const v8bf*)(p + 16);
    return f.v;
}
__device__ __forceinline__ v8f wmma_bf(v16bf a, v16bf b, v8f c) {
    return __builtin_amdgcn_wmma_f32_16x16x32_bf16(false, a, false, b, (short)0, c, false, false);
}

// CDNA5 async copy: 16B per lane, global -> LDS, tracked by ASYNCcnt.
__device__ __forceinline__ void async_b128(unsigned lds_off, const bf16_t* g) {
    unsigned long long ga = (unsigned long long)g;
    asm volatile("global_load_async_to_lds_b128 %0, %1, off"
                 :: "v"(lds_off), "v"(ga) : "memory");
}
__device__ __forceinline__ void wait_async0() {
    asm volatile("s_wait_asynccnt 0x0" ::: "memory");
}
__device__ __forceinline__ unsigned lds_addr(const void* p) {
    return (unsigned)(size_t)p;      // LDS aperture keeps offset in addr[31:0]
}

// ---------------------------------------------------------------- sizes
#define DMODEL 2048
#define SEQ    2048
#define TOK    4096          // BATCH*SEQ
#define NPROJ_PAD 4736       // 4640 padded to 37*128
#define CONVD  2560
#define NSTATE 256
#define NHEAD  32
#define HDIM   64
#define NCHUNK 64            // BATCH * (SEQ/64)

// ---------------------------------------------------------------- f32 -> bf16 (row-major, pad cols)
__global__ void cvt_pad_kernel(const float* __restrict__ src, bf16_t* __restrict__ dst,
                               int rows, int cols, int ldd) {
    size_t idx = (size_t)blockIdx.x * 256 + threadIdx.x;
    size_t total = (size_t)rows * ldd;
    if (idx >= total) return;
    int c = (int)(idx % ldd);
    size_t r = idx / ldd;
    dst[idx] = (c < cols) ? to_bf16(src[r * (size_t)cols + c]) : to_bf16(0.f);
}

// ---------------------------------------------------------------- f32 [K][N] -> bf16 [Npad][K] (weightsᵀ)
__global__ __launch_bounds__(256) void transpose_cvt_kernel(
    const float* __restrict__ src, bf16_t* __restrict__ dst, int K, int N) {
    __shared__ float tile[32][33];
    const int n0 = blockIdx.x * 32, k0 = blockIdx.y * 32;
    const int tx = threadIdx.x & 31, ty = threadIdx.x >> 5;   // 32 x 8
    #pragma unroll
    for (int i = 0; i < 32; i += 8) {
        int n = n0 + tx;
        tile[ty + i][tx] = (n < N) ? src[(size_t)(k0 + ty + i) * N + n] : 0.f;
    }
    __syncthreads();
    #pragma unroll
    for (int i = 0; i < 32; i += 8)
        dst[(size_t)(n0 + ty + i) * K + k0 + tx] = to_bf16(tile[tx][ty + i]);
}

// ---------------------------------------------------------------- rmsnorm -> bf16
__global__ __launch_bounds__(256) void rmsnorm_kernel(const float* __restrict__ x,
                                                      const float* __restrict__ w,
                                                      bf16_t* __restrict__ out) {
    const int t = blockIdx.x, tid = threadIdx.x;
    const float* xr = x + (size_t)t * DMODEL;
    __shared__ float red[256];
    float s = 0.f;
    for (int i = tid; i < DMODEL; i += 256) { float v = xr[i]; s += v * v; }
    red[tid] = s; __syncthreads();
    for (int o = 128; o > 0; o >>= 1) { if (tid < o) red[tid] += red[tid + o]; __syncthreads(); }
    float scale = rsqrtf(red[0] * (1.f / DMODEL) + 1e-5f);
    for (int i = tid; i < DMODEL; i += 256)
        out[(size_t)t * DMODEL + i] = to_bf16(xr[i] * scale * w[i]);
}

// ---------------------------------------------------------------- async double-buffered bf16 GEMM
// C[M][ldc] = A[M][lda] x BT[N][ldbt]^T.  Block 128x128, BK=32, 8 waves, 32x64 per wave.
__global__ __launch_bounds__(256) void gemm_async_kernel(
    const bf16_t* __restrict__ A, const bf16_t* __restrict__ BT, float* __restrict__ C,
    int M, int N, int K, int lda, int ldbt, int ldc) {
    __shared__ __align__(16) bf16_t sA[2][128 * 40];
    __shared__ __align__(16) bf16_t sB[2][128 * 40];
    const int tid = threadIdx.x, lane = tid & 31, wv = tid >> 5;
    const int wm = wv >> 1, wn = wv & 1;                 // 4(M) x 2(N) waves
    const int bm = blockIdx.y * 128, bn = blockIdx.x * 128;
    const int r0 = tid >> 2, kc0 = (tid & 3) * 8;        // 16B chunk per thread (x2 rows)

    const bf16_t* gA = A + (size_t)(bm + r0) * lda + kc0;
    const bf16_t* gB = BT + (size_t)(bn + r0) * ldbt + kc0;
    const unsigned dA0[2] = { lds_addr(&sA[0][r0 * 40 + kc0]), lds_addr(&sA[1][r0 * 40 + kc0]) };
    const unsigned dB0[2] = { lds_addr(&sB[0][r0 * 40 + kc0]), lds_addr(&sB[1][r0 * 40 + kc0]) };

    v8f acc[2][4] = {};
    // prologue: tile 0 into buffer 0
    async_b128(dA0[0], gA);
    async_b128(dA0[0] + 64 * 40 * 2, gA + (size_t)64 * lda);
    async_b128(dB0[0], gB);
    async_b128(dB0[0] + 64 * 40 * 2, gB + (size_t)64 * ldbt);

    int buf = 0;
    for (int k0 = 0; k0 < K; k0 += 32) {
        wait_async0();            // this wave's chunks of tile k0 are in LDS
        __syncthreads();          // everyone's chunks visible; prev reads of buf^1 done
        if (k0 + 32 < K) {        // prefetch next tile into other buffer
            int nb = buf ^ 1;
            async_b128(dA0[nb], gA + k0 + 32);
            async_b128(dA0[nb] + 64 * 40 * 2, gA + (size_t)64 * lda + k0 + 32);
            async_b128(dB0[nb], gB + k0 + 32);
            async_b128(dB0[nb] + 64 * 40 * 2, gB + (size_t)64 * ldbt + k0 + 32);
        }
        const bf16_t* pa = &sA[buf][(wm * 32) * 40];
        const bf16_t* pb = &sB[buf][(wn * 64) * 40];
        v16bf fa0 = load_frag_a(pa, 40, lane);
        v16bf fa1 = load_frag_a(pa + 16 * 40, 40, lane);
        v16bf fb0 = load_frag_b(pb, 40, lane);
        v16bf fb1 = load_frag_b(pb + 16 * 40, 40, lane);
        v16bf fb2 = load_frag_b(pb + 32 * 40, 40, lane);
        v16bf fb3 = load_frag_b(pb + 48 * 40, 40, lane);
        acc[0][0] = wmma_bf(fa0, fb0, acc[0][0]);
        acc[0][1] = wmma_bf(fa0, fb1, acc[0][1]);
        acc[0][2] = wmma_bf(fa0, fb2, acc[0][2]);
        acc[0][3] = wmma_bf(fa0, fb3, acc[0][3]);
        acc[1][0] = wmma_bf(fa1, fb0, acc[1][0]);
        acc[1][1] = wmma_bf(fa1, fb1, acc[1][1]);
        acc[1][2] = wmma_bf(fa1, fb2, acc[1][2]);
        acc[1][3] = wmma_bf(fa1, fb3, acc[1][3]);
        buf ^= 1;
    }
    const int half = lane >> 4, lr = lane & 15;
    #pragma unroll
    for (int mi = 0; mi < 2; ++mi)
        #pragma unroll
        for (int ni = 0; ni < 4; ++ni) {
            int row0 = bm + wm * 32 + mi * 16 + half * 8;
            int col  = bn + wn * 64 + ni * 16 + lr;
            #pragma unroll
            for (int r = 0; r < 8; ++r)
                C[(size_t)(row0 + r) * ldc + col] = acc[mi][ni][r];
        }
}

// ---------------------------------------------------------------- dt = softplus(dt + bias)
__global__ void dt_kernel(const float* __restrict__ proj, const float* __restrict__ dt_bias,
                          float* __restrict__ dtv) {
    int idx = blockIdx.x * 256 + threadIdx.x;
    if (idx >= TOK * NHEAD) return;
    int t = idx >> 5, h = idx & 31;
    float v = proj[(size_t)t * NPROJ_PAD + 4608 + h] + dt_bias[h];
    dtv[idx] = (v > 20.f) ? v : log1pf(expf(v));
}

// ---------------------------------------------------------------- per-chunk cumsum of dt*A
__global__ void cumsum_kernel(const float* __restrict__ dtv, const float* __restrict__ A_log,
                              float* __restrict__ acs) {
    int idx = blockIdx.x * 256 + threadIdx.x;
    if (idx >= NCHUNK * NHEAD) return;
    int gc = idx >> 5, h = idx & 31;
    float A = -expf(A_log[h]);
    float s = 0.f;
    for (int l = 0; l < 64; ++l) {
        size_t o = ((size_t)gc * 64 + l) * NHEAD + h;
        s += dtv[o] * A;
        acs[o] = s;
    }
}

// ---------------------------------------------------------------- causal conv(K=4) + silu
__global__ void conv_kernel(const float* __restrict__ proj, const float* __restrict__ conv_w,
                            const float* __restrict__ conv_b, float* __restrict__ convo) {
    size_t idx = (size_t)blockIdx.x * 256 + threadIdx.x;
    if (idx >= (size_t)TOK * CONVD) return;
    int j = (int)(idx % CONVD);
    int t = (int)(idx / CONVD);
    int l = t & (SEQ - 1);
    const float* wc = conv_w + (size_t)j * 4;
    float acc = conv_b[j];
    #pragma unroll
    for (int k = 0; k < 4; ++k) {
        int src = l - 3 + k;
        if (src >= 0) acc += proj[(size_t)(t - 3 + k) * NPROJ_PAD + DMODEL + j] * wc[k];
    }
    acc = acc / (1.f + expf(-acc));
    convo[idx] = acc;
}

// ---------------------------------------------------------------- per-chunk states = (xs*dt*decay)^T @ B   (WMMA)
__global__ __launch_bounds__(256) void chunk_states_kernel(
    const float* __restrict__ convo, const float* __restrict__ dtv,
    const float* __restrict__ acs, float* __restrict__ states) {
    const int gc = blockIdx.x, h = blockIdx.y;
    __shared__ __align__(16) bf16_t sXwT[64 * 72];    // [p][l] = xs[l][p]*w[l]
    __shared__ __align__(16) bf16_t sBT[256 * 72];    // [n][l] = B[l][n]
    __shared__ float sW[64];
    const int tid = threadIdx.x, t0 = gc * 64;
    if (tid < 64) {
        float last = acs[((size_t)t0 + 63) * NHEAD + h];
        sW[tid] = dtv[((size_t)t0 + tid) * NHEAD + h] *
                  expf(last - acs[((size_t)t0 + tid) * NHEAD + h]);
    }
    __syncthreads();
    for (int i = tid; i < 64 * 64; i += 256) {
        int l = i >> 6, n0 = (i & 63) * 4;
        float4 v = *(const float4*)(convo + (size_t)(t0 + l) * CONVD + DMODEL + n0);
        sBT[(n0 + 0) * 72 + l] = to_bf16(v.x);
        sBT[(n0 + 1) * 72 + l] = to_bf16(v.y);
        sBT[(n0 + 2) * 72 + l] = to_bf16(v.z);
        sBT[(n0 + 3) * 72 + l] = to_bf16(v.w);
    }
    for (int i = tid; i < 64 * 16; i += 256) {
        int l = i >> 4, p0 = (i & 15) * 4;
        float4 v = *(const float4*)(convo + (size_t)(t0 + l) * CONVD + h * HDIM + p0);
        float wl = sW[l];
        sXwT[(p0 + 0) * 72 + l] = to_bf16(v.x * wl);
        sXwT[(p0 + 1) * 72 + l] = to_bf16(v.y * wl);
        sXwT[(p0 + 2) * 72 + l] = to_bf16(v.z * wl);
        sXwT[(p0 + 3) * 72 + l] = to_bf16(v.w * wl);
    }
    __syncthreads();
    const int lane = tid & 31, w = tid >> 5;
    const int tm = w >> 1;               // p tile 0..3
    const int tn0 = (w & 1) * 8;         // 8 n tiles per wave
    v8f acc[8] = {};
    for (int k0 = 0; k0 < 64; k0 += 32) {
        v16bf fa = load_frag_a(sXwT + (tm * 16) * 72 + k0, 72, lane);
        #pragma unroll
        for (int j = 0; j < 8; ++j) {
            v16bf fb = load_frag_b(sBT + ((tn0 + j) * 16) * 72 + k0, 72, lane);
            acc[j] = wmma_bf(fa, fb, acc[j]);
        }
    }
    const int half = lane >> 4, lr = lane & 15;
    float* base = states + ((size_t)gc * NHEAD + h) * (HDIM * NSTATE);
    #pragma unroll
    for (int j = 0; j < 8; ++j)
        #pragma unroll
        for (int r = 0; r < 8; ++r) {
            int p = tm * 16 + half * 8 + r;
            int n = (tn0 + j) * 16 + lr;
            base[(size_t)p * NSTATE + n] = acc[j][r];
        }
}

// ---------------------------------------------------------------- sequential inter-chunk scan (in place: states -> prev)
__global__ __launch_bounds__(256) void scan_kernel(const float* __restrict__ acs,
                                                   float* __restrict__ states) {
    const int b = blockIdx.x >> 5, h = blockIdx.x & 31;
    const int tid = threadIdx.x;
    float4 S[16];
    #pragma unroll
    for (int i = 0; i < 16; ++i) S[i] = make_float4(0.f, 0.f, 0.f, 0.f);
    for (int c = 0; c < 32; ++c) {
        int gc = b * 32 + c;
        float dec = expf(acs[((size_t)gc * 64 + 63) * NHEAD + h]);
        float4* base = (float4*)(states + ((size_t)gc * NHEAD + h) * (HDIM * NSTATE));
        #pragma unroll
        for (int i = 0; i < 16; ++i) {
            float4* p = base + (size_t)i * 256 + tid;
            float4 st = *p;
            *p = S[i];
            S[i].x = S[i].x * dec + st.x;
            S[i].y = S[i].y * dec + st.y;
            S[i].z = S[i].z * dec + st.z;
            S[i].w = S[i].w * dec + st.w;
        }
    }
}

// ---------------------------------------------------------------- Y = (CB*L) @ xt + exp(Acs)*(C @ prev^T) + D*xs  (WMMA)
__global__ __launch_bounds__(256) void chunk_out_kernel(
    const float* __restrict__ convo, const bf16_t* __restrict__ convbf,
    const float* __restrict__ dtv, const float* __restrict__ acs,
    const float* __restrict__ prev, const float* __restrict__ Dskip,
    float* __restrict__ y) {
    const int gc = blockIdx.x, h = blockIdx.y;
    __shared__ __align__(16) bf16_t sPb[64 * 264];    // prev  [p][n] bf16
    __shared__ __align__(16) bf16_t sXTb[64 * 72];    // xs^T  [p][l]
    __shared__ __align__(16) bf16_t sMb[64 * 72];     // masked/decayed CB [l][s]
    __shared__ float sA[64], sDt[64];
    const int tid = threadIdx.x, t0 = gc * 64;
    const int lane = tid & 31, w = tid >> 5;
    if (tid < 64) {
        sA[tid]  = acs[((size_t)t0 + tid) * NHEAD + h];
        sDt[tid] = dtv[((size_t)t0 + tid) * NHEAD + h];
    }
    const float* pbase = prev + ((size_t)gc * NHEAD + h) * (HDIM * NSTATE);
    for (int i = tid; i < 4096; i += 256) {
        int p = i >> 6, n0 = (i & 63) * 4;
        float4 v = *(const float4*)(pbase + (size_t)p * NSTATE + n0);
        bf16_t* d = sPb + p * 264 + n0;
        d[0] = to_bf16(v.x); d[1] = to_bf16(v.y); d[2] = to_bf16(v.z); d[3] = to_bf16(v.w);
    }
    for (int i = tid; i < 4096; i += 256) {
        int l = i >> 6, p = i & 63;
        sXTb[p * 72 + l] = convbf[(size_t)(t0 + l) * CONVD + h * HDIM + p];
    }
    const int tm = w >> 1;            // l tile 0..3
    const int tn0 = (w & 1) * 2;      // s/p tile pair
    const bf16_t* Cg = convbf + (size_t)t0 * CONVD + DMODEL + NSTATE;
    const bf16_t* Bg = convbf + (size_t)t0 * CONVD + DMODEL;
    v8f cb0 = {}, cb1 = {};
    for (int k0 = 0; k0 < 256; k0 += 32) {
        v16bf fa = load_frag_a(Cg + (size_t)(tm * 16) * CONVD + k0, CONVD, lane);
        v16bf f0 = load_frag_b(Bg + (size_t)(tn0 * 16) * CONVD + k0, CONVD, lane);
        v16bf f1 = load_frag_b(Bg + (size_t)((tn0 + 1) * 16) * CONVD + k0, CONVD, lane);
        cb0 = wmma_bf(fa, f0, cb0);
        cb1 = wmma_bf(fa, f1, cb1);
    }
    __syncthreads();
    const int half = lane >> 4, lr = lane & 15;
    #pragma unroll
    for (int r = 0; r < 8; ++r) {
        int l = tm * 16 + half * 8 + r;
        int s0 = tn0 * 16 + lr, s1 = (tn0 + 1) * 16 + lr;
        float m0 = (s0 <= l) ? cb0[r] * expf(sA[l] - sA[s0]) * sDt[s0] : 0.f;
        float m1 = (s1 <= l) ? cb1[r] * expf(sA[l] - sA[s1]) * sDt[s1] : 0.f;
        sMb[l * 72 + s0] = to_bf16(m0);
        sMb[l * 72 + s1] = to_bf16(m1);
    }
    __syncthreads();
    v8f y0 = {}, y1 = {}, d0 = {}, d1 = {};
    for (int k0 = 0; k0 < 256; k0 += 32) {
        v16bf fa = load_frag_a(Cg + (size_t)(tm * 16) * CONVD + k0, CONVD, lane);
        v16bf f0 = load_frag_b(sPb + (tn0 * 16) * 264 + k0, 264, lane);
        v16bf f1 = load_frag_b(sPb + ((tn0 + 1) * 16) * 264 + k0, 264, lane);
        y0 = wmma_bf(fa, f0, y0);
        y1 = wmma_bf(fa, f1, y1);
    }
    for (int k0 = 0; k0 < 64; k0 += 32) {
        v16bf fa = load_frag_a(sMb + (tm * 16) * 72 + k0, 72, lane);
        v16bf f0 = load_frag_b(sXTb + (tn0 * 16) * 72 + k0, 72, lane);
        v16bf f1 = load_frag_b(sXTb + ((tn0 + 1) * 16) * 72 + k0, 72, lane);
        d0 = wmma_bf(fa, f0, d0);
        d1 = wmma_bf(fa, f1, d1);
    }
    const float dsk = Dskip[h];
    #pragma unroll
    for (int r = 0; r < 8; ++r) {
        int l = tm * 16 + half * 8 + r;
        float eA = expf(sA[l]);
        int t = t0 + l;
        int p0 = tn0 * 16 + lr, p1 = (tn0 + 1) * 16 + lr;
        float xs0 = convo[(size_t)t * CONVD + h * HDIM + p0];
        float xs1 = convo[(size_t)t * CONVD + h * HDIM + p1];
        y[(size_t)t * DMODEL + h * HDIM + p0] = y0[r] * eA + d0[r] + dsk * xs0;
        y[(size_t)t * DMODEL + h * HDIM + p1] = y1[r] * eA + d1[r] + dsk * xs1;
    }
}

// ---------------------------------------------------------------- gated rmsnorm -> bf16
__global__ __launch_bounds__(256) void gatenorm_kernel(const float* __restrict__ y,
                                                       const float* __restrict__ proj,
                                                       const float* __restrict__ gw,
                                                       bf16_t* __restrict__ out) {
    const int t = blockIdx.x, tid = threadIdx.x;
    const float* yr = y + (size_t)t * DMODEL;
    const float* zr = proj + (size_t)t * NPROJ_PAD;   // z = proj[:, :2048]
    __shared__ float red[256];
    float s = 0.f;
    for (int i = tid; i < DMODEL; i += 256) {
        float z = zr[i];
        float v = yr[i] * (z / (1.f + expf(-z)));
        s += v * v;
    }
    red[tid] = s; __syncthreads();
    for (int o = 128; o > 0; o >>= 1) { if (tid < o) red[tid] += red[tid + o]; __syncthreads(); }
    float scale = rsqrtf(red[0] * (1.f / DMODEL) + 1e-5f);
    for (int i = tid; i < DMODEL; i += 256) {
        float z = zr[i];
        float v = yr[i] * (z / (1.f + expf(-z)));
        out[(size_t)t * DMODEL + i] = to_bf16(v * scale * gw[i]);
    }
}

// ---------------------------------------------------------------- workspace layout (bytes)
static const size_t OFF_WIN_BF  = 0;                       // bf16 4736x2048 (w_inᵀ)
static const size_t OFF_WOUT_BF = 19398656;                // bf16 2048x2048 (w_outᵀ)
static const size_t OFF_WO_BF   = 27787264;                // bf16 2048x2048 (wo_wᵀ)
static const size_t OFF_XN      = 36175872;                // bf16 4096x2048 (later: tmp bf16)
static const size_t OFF_PROJ    = 52953088;                // f32  4096x4736 (later: tmp f32)
static const size_t OFF_CONV    = 130547712;               // f32  4096x2560
static const size_t OFF_CONVBF  = 172490752;               // bf16 4096x2560
static const size_t OFF_DTV     = 193462272;               // f32  4096x32
static const size_t OFF_ACS     = 193986560;               // f32  4096x32
static const size_t OFF_STATES  = 194510848;               // f32  64x32x64x256 (later: yn bf16)
static const size_t OFF_Y       = 328728576;               // f32  4096x2048

extern "C" void kernel_launch(void* const* d_in, const int* in_sizes, int n_in,
                              void* d_out, int out_size, void* d_ws, size_t ws_size,
                              hipStream_t stream) {
    (void)in_sizes; (void)n_in; (void)out_size; (void)ws_size;
    const float* x       = (const float*)d_in[0];
    const float* norm_w  = (const float*)d_in[1];
    const float* w_in    = (const float*)d_in[2];
    const float* conv_w  = (const float*)d_in[3];
    const float* conv_b  = (const float*)d_in[4];
    const float* dt_bias = (const float*)d_in[5];
    const float* A_log   = (const float*)d_in[6];
    const float* D_skip  = (const float*)d_in[7];
    const float* gnorm_w = (const float*)d_in[8];
    const float* w_out   = (const float*)d_in[9];
    const float* wo_w    = (const float*)d_in[10];
    float* out = (float*)d_out;
    char* ws = (char*)d_ws;

    bf16_t* winbf  = (bf16_t*)(ws + OFF_WIN_BF);
    bf16_t* woutbf = (bf16_t*)(ws + OFF_WOUT_BF);
    bf16_t* wobf   = (bf16_t*)(ws + OFF_WO_BF);
    bf16_t* xnbf   = (bf16_t*)(ws + OFF_XN);
    float*  projf  = (float*)(ws + OFF_PROJ);
    float*  convf  = (float*)(ws + OFF_CONV);
    bf16_t* convbf = (bf16_t*)(ws + OFF_CONVBF);
    float*  dtv    = (float*)(ws + OFF_DTV);
    float*  acsf   = (float*)(ws + OFF_ACS);
    float*  statef = (float*)(ws + OFF_STATES);
    float*  yf     = (float*)(ws + OFF_Y);
    bf16_t* ynbf   = (bf16_t*)(ws + OFF_STATES);   // alias: states dead after chunk_out
    float*  tmpf   = (float*)(ws + OFF_PROJ);      // alias: proj dead after gatenorm
    bf16_t* tmpbf  = (bf16_t*)(ws + OFF_XN);       // alias: xn dead after gemm1

    // weights -> transposed bf16 [N][K] (w_in zero-padded to N=4736)
    transpose_cvt_kernel<<<dim3(NPROJ_PAD / 32, 2048 / 32), 256, 0, stream>>>(w_in, winbf, 2048, 4640);
    transpose_cvt_kernel<<<dim3(64, 64), 256, 0, stream>>>(w_out, woutbf, 2048, 2048);
    transpose_cvt_kernel<<<dim3(64, 64), 256, 0, stream>>>(wo_w, wobf, 2048, 2048);

    rmsnorm_kernel<<<TOK, 256, 0, stream>>>(x, norm_w, xnbf);

    // proj = xn @ w_in   (M=4096, N=4736, K=2048)
    gemm_async_kernel<<<dim3(NPROJ_PAD / 128, TOK / 128), 256, 0, stream>>>(
        xnbf, winbf, projf, TOK, NPROJ_PAD, 2048, 2048, 2048, NPROJ_PAD);

    dt_kernel<<<(TOK * NHEAD) / 256, 256, 0, stream>>>(projf, dt_bias, dtv);
    cumsum_kernel<<<(NCHUNK * NHEAD + 255) / 256, 256, 0, stream>>>(dtv, A_log, acsf);
    conv_kernel<<<(TOK * CONVD) / 256, 256, 0, stream>>>(projf, conv_w, conv_b, convf);
    cvt_pad_kernel<<<(TOK * CONVD) / 256, 256, 0, stream>>>(convf, convbf, TOK, CONVD, CONVD);

    chunk_states_kernel<<<dim3(NCHUNK, NHEAD), 256, 0, stream>>>(convf, dtv, acsf, statef);
    scan_kernel<<<64, 256, 0, stream>>>(acsf, statef);
    chunk_out_kernel<<<dim3(NCHUNK, NHEAD), 256, 0, stream>>>(convf, convbf, dtv, acsf,
                                                              statef, D_skip, yf);

    gatenorm_kernel<<<TOK, 256, 0, stream>>>(yf, projf, gnorm_w, ynbf);

    // out = ((yn @ w_out) @ wo_w)
    gemm_async_kernel<<<dim3(2048 / 128, TOK / 128), 256, 0, stream>>>(
        ynbf, woutbf, tmpf, TOK, 2048, 2048, 2048, 2048, 2048);
    cvt_pad_kernel<<<(TOK * 2048) / 256, 256, 0, stream>>>(tmpf, tmpbf, TOK, 2048, 2048);
    gemm_async_kernel<<<dim3(2048 / 128, TOK / 128), 256, 0, stream>>>(
        tmpbf, wobf, out, TOK, 2048, 2048, 2048, 2048, 2048);
}
